// GraphTopologicalRegularization_56573309223817
// MI455X (gfx1250) — compile-verified, matrix operand order checked by reading, not run on hardware
//
#include <hip/hip_runtime.h>
#include <hip/hip_bf16.h>

typedef __attribute__((ext_vector_type(16))) _Float16 v16h;
typedef __attribute__((ext_vector_type(8)))  float    v8f;

// ---------------------------------------------------------------------------
// Connected-components (Betti-0) via min-label hooking + pointer jumping.
// loss = betti0 (betti1 == 0, weight == 1). node_features is unused by the
// reference output, so we never touch d_in[0].
// ---------------------------------------------------------------------------

__global__ void cc_zero_out_kernel(float* __restrict__ out) {
    if (blockIdx.x == 0 && threadIdx.x == 0) out[0] = 0.0f;
}

__global__ void cc_init_labels_kernel(int* __restrict__ labels, int n) {
    int i = blockIdx.x * blockDim.x + threadIdx.x;
    if (i < n) labels[i] = i;
}

// Hooking pass: for each (undirected) edge pull the larger endpoint label
// down to the smaller one. Monotone-decreasing atomicMin => deterministic
// fixed point (component min-label). Edge arrays stream coalesced; prefetch
// ahead (speculative global_prefetch_b8 — OOB prefetches are silently
// dropped by hardware).
__global__ void cc_hook_edges_kernel(const int* __restrict__ src,
                                     const int* __restrict__ dst,
                                     int* __restrict__ labels, int nE) {
    int i = blockIdx.x * blockDim.x + threadIdx.x;
    if (i >= nE) return;
    __builtin_prefetch(&src[i + 16384], 0, 0);
    __builtin_prefetch(&dst[i + 16384], 0, 0);
    int u = src[i];
    int v = dst[i];
    int lu = labels[u];
    int lv = labels[v];
    if (lu < lv)      atomicMin(&labels[v], lu);
    else if (lv < lu) atomicMin(&labels[u], lv);
}

// Pointer jumping: nxt[i] = cur[cur[i]] (double-buffered, race-free).
__global__ void cc_pointer_jump_kernel(const int* __restrict__ cur,
                                       int* __restrict__ nxt, int n) {
    int i = blockIdx.x * blockDim.x + threadIdx.x;
    if (i < n) nxt[i] = cur[cur[i]];
}

// Root count = number of components. 512 threads/block accumulate integer
// indicators (each <= a few, exact in f16), then wave 0 collapses all 512
// partials with ONE v_wmma_f32_16x16x32_f16 against an all-ones B matrix:
// D[m][n] = sum_k A[m][k]. Lane0 holds rows 0..7 in c[0..7], lane16 rows
// 8..15; one per-lane add + one 16-lane shuffle finishes the block sum.
// Integer-valued f32 adds are exact => atomicAdd order doesn't matter.
__global__ __launch_bounds__(512)
void cc_count_roots_kernel(const int* __restrict__ labels, int n,
                           float* __restrict__ out) {
    __shared__ __align__(32) _Float16 scnt[512];
    const int tid = threadIdx.x;
    int local = 0;
    for (int i = blockIdx.x * blockDim.x + tid; i < n;
         i += gridDim.x * blockDim.x)
        local += (labels[i] == i) ? 1 : 0;
    scnt[tid] = (_Float16)local;
    __syncthreads();

    if (tid < 32) {  // wave 0 only: EXEC is all-ones inside this branch
        v16h a = *(const v16h*)(&scnt[tid * 16]);
        v16h b;
#pragma unroll
        for (int k = 0; k < 16; ++k) b[k] = (_Float16)1.0f;
        v8f c = {};
        // D = A * ones + 0  => per-row sums of the 16x32 partial-count tile
        c = __builtin_amdgcn_wmma_f32_16x16x32_f16(
                /*neg_a=*/false, a, /*neg_b=*/false, b,
                /*c_mod=*/(short)0, c, /*reuse_a=*/false, /*reuse_b=*/false);
        float s = c[0] + c[1] + c[2] + c[3] + c[4] + c[5] + c[6] + c[7];
        s += __shfl_down(s, 16, 32);   // add rows 8..15 (lane 16) into lane 0
        if (tid == 0) atomicAdd(out, s);
    }
}

extern "C" void kernel_launch(void* const* d_in, const int* in_sizes, int n_in,
                              void* d_out, int out_size, void* d_ws, size_t ws_size,
                              hipStream_t stream) {
    (void)n_in; (void)out_size; (void)ws_size;

    const int D_FEAT = 256;
    const int N = in_sizes[0] / D_FEAT;          // 50000 nodes
    const int E = in_sizes[1] / 2;               // 1.6M edges
    const int* edge_index = (const int*)d_in[1]; // (2, E) row-major
    const int* src = edge_index;
    const int* dst = edge_index + E;
    float* out = (float*)d_out;

    // Scratch layout: two label buffers (double buffer for pointer jumping).
    size_t lab_bytes = (((size_t)N * sizeof(int)) + 255) & ~(size_t)255;
    int* labA = (int*)d_ws;
    int* labB = (int*)((char*)d_ws + lab_bytes);

    cc_zero_out_kernel<<<1, 64, 0, stream>>>(out);
    cc_init_labels_kernel<<<(N + 255) / 256, 256, 0, stream>>>(labA, N);

    int* cur = labA;
    int* nxt = labB;
    const int N_ITERS = 20;  // ceil(log2(50000)) + 4, matches reference budget
    for (int it = 0; it < N_ITERS; ++it) {
        cc_hook_edges_kernel<<<(E + 255) / 256, 256, 0, stream>>>(src, dst, cur, E);
        cc_pointer_jump_kernel<<<(N + 255) / 256, 256, 0, stream>>>(cur, nxt, N);
        int* t = cur; cur = nxt; nxt = t;
    }

    cc_count_roots_kernel<<<64, 512, 0, stream>>>(cur, N, out);
}
